// multihead_attentive_layer_45286135169420
// MI455X (gfx1250) — matrix operand order repaired and checked by reading, not compiled, not run on hardware
//
#include <hip/hip_runtime.h>

// ---------------------------------------------------------------------------
// Problem constants (from the reference)
// ---------------------------------------------------------------------------
#define B_   8
#define SQ_  2048
#define SKV_ 2048
#define D_   1024
#define H_   8
#define DH_  128
#define P_   1024
#define HD_  (H_ * DH_)   // 1024
#define LN_EPS 1e-3f

typedef __bf16 bf16;
typedef __attribute__((ext_vector_type(16))) __bf16 v16bf;
typedef __attribute__((ext_vector_type(8)))  __bf16 v8bf;
typedef __attribute__((ext_vector_type(8)))  float  v8f;
typedef __attribute__((ext_vector_type(4)))  int    v4i;

// CDNA5 async global->LDS copy (ASYNCcnt path), with plain-load fallback.
#if __has_builtin(__builtin_amdgcn_global_load_async_to_lds_b128)
#define ASYNC_LDS 1
#else
#define ASYNC_LDS 0
#endif

// ---------------------------------------------------------------------------
// Helpers
// ---------------------------------------------------------------------------
__device__ __forceinline__ bf16 f2bf(float f) {
  union { float f; unsigned u; } x; x.f = f;
  unsigned r = x.u + 0x7FFFu + ((x.u >> 16) & 1u);   // round-to-nearest-even
  union { unsigned short u; bf16 b; } y; y.u = (unsigned short)(r >> 16);
  return y.b;
}

__device__ __forceinline__ v8f v8zero() {
  v8f a;
#pragma unroll
  for (int i = 0; i < 8; ++i) a[i] = 0.0f;
  return a;
}

// 16-byte global -> LDS copy, async when the toolchain exposes the builtin.
__device__ __forceinline__ void cp_g2l_16B(const bf16* g, bf16* l) {
#if ASYNC_LDS
  __builtin_amdgcn_global_load_async_to_lds_b128(
      (v4i __attribute__((address_space(1)))*)g,
      (v4i __attribute__((address_space(3)))*)l, 0, 0);
#else
  *(v8bf*)l = *(const v8bf*)g;
#endif
}

__device__ __forceinline__ void wait_async_lds() {
#if ASYNC_LDS
#if __has_builtin(__builtin_amdgcn_s_wait_asynccnt)
  __builtin_amdgcn_s_wait_asynccnt(0);
#else
  asm volatile("s_wait_asynccnt 0x0" ::: "memory");
#endif
#endif
}

// A fragment: 16x32 bf16, rows striped in lanes (ISA 7.12.2).
// lane L: row = m0 + (L&15); holds K = kc..kc+7 and kc+16..kc+23, kc = (L>=16)?8:0
__device__ __forceinline__ v16bf load_a_frag(const bf16* __restrict__ A, int lda,
                                             int m0, int k0) {
  const int lane = threadIdx.x & 31;
  const int m  = m0 + (lane & 15);
  const int kc = k0 + ((lane >> 4) << 3);
  const bf16* p = A + (size_t)m * lda + kc;
  v8bf lo = *(const v8bf*)(p);
  v8bf hi = *(const v8bf*)(p + 16);
  return __builtin_shufflevector(lo, hi, 0,1,2,3,4,5,6,7,8,9,10,11,12,13,14,15);
}

// B fragment: 32x16 bf16 (KxN), loaded from BT stored row-major [N][K].
// lane L: col = n0 + (L&15); holds K = kc..kc+7 and kc+16..kc+23, kc = k0+((L>=16)?8:0)
__device__ __forceinline__ v16bf load_b_frag(const bf16* __restrict__ BT, int ldb,
                                             int n0, int k0) {
  const int lane = threadIdx.x & 31;
  const int n  = n0 + (lane & 15);
  const int kc = k0 + ((lane >> 4) << 3);
  const bf16* p = BT + (size_t)n * ldb + kc;
  v8bf lo = *(const v8bf*)(p);
  v8bf hi = *(const v8bf*)(p + 16);
  return __builtin_shufflevector(lo, hi, 0,1,2,3,4,5,6,7,8,9,10,11,12,13,14,15);
}

__device__ __forceinline__ v8f wmma_bf16(v16bf a, v16bf b, v8f c) {
  return __builtin_amdgcn_wmma_f32_16x16x32_bf16(
      /*neg_a=*/false, a, /*neg_b=*/false, b,
      /*c_mod=*/(short)0, c, /*reuse_a=*/false, /*reuse_b=*/false);
}

// ---------------------------------------------------------------------------
// f32 -> bf16 convert (grid-stride)
// ---------------------------------------------------------------------------
__global__ void __launch_bounds__(256)
cvt_kernel(const float* __restrict__ in, bf16* __restrict__ out, size_t n) {
  for (size_t i = (size_t)blockIdx.x * 256 + threadIdx.x; i < n;
       i += (size_t)gridDim.x * 256)
    out[i] = f2bf(in[i]);
}

// f32 [R][C] -> bf16 transposed [C][R]; blockIdx.z selects matrix slice
__global__ void __launch_bounds__(256)
tcvt_kernel(const float* __restrict__ in, bf16* __restrict__ out, int R, int C) {
  const size_t total = (size_t)R * C;
  const float* ip = in  + (size_t)blockIdx.z * total;
  bf16*        op = out + (size_t)blockIdx.z * total;
  for (size_t idx = (size_t)blockIdx.x * 256 + threadIdx.x; idx < total;
       idx += (size_t)gridDim.x * 256) {
    int r = (int)(idx / C), c = (int)(idx % C);
    op[(size_t)c * R + r] = f2bf(ip[idx]);
  }
}

// ---------------------------------------------------------------------------
// Batched per-head projection GEMM: out[z] = X[b] @ WT[h]^T
//   X  : bf16 [B][S=2048][D]         (A, row-major, lda = D)
//   WT : bf16 [H][DH][D]             (B^T, row-major, ldb = D)
//   out: bf16 [B*H][S][DH]           (transposed==0)
//        bf16 [B*H][DH][S]           (transposed==1, used for V)
// grid (S/128, DH/128, B*H), block 256: 8 waves as 4(M)x2(N), 32x64 per wave
// ---------------------------------------------------------------------------
__global__ void __launch_bounds__(256)
gemm_qkv_kernel(const bf16* __restrict__ X, const bf16* __restrict__ WT,
                bf16* __restrict__ out, int transposed) {
  const int lane = threadIdx.x & 31;
  const int w    = threadIdx.x >> 5;
  const int wm = w & 3, wn = w >> 2;
  const int z = blockIdx.z, b = z / H_, h = z % H_;
  const int m0 = blockIdx.x * 128 + wm * 32;
  const int n0 = blockIdx.y * 128 + wn * 64;

  const bf16* A  = X  + (size_t)b * SQ_ * D_;
  const bf16* BT = WT + (size_t)h * DH_ * D_;

  v8f acc[2][4];
#pragma unroll
  for (int i = 0; i < 2; ++i)
#pragma unroll
    for (int j = 0; j < 4; ++j) acc[i][j] = v8zero();

  for (int k = 0; k < D_; k += 32) {
    if (k + 32 < D_) {
      __builtin_prefetch(A  + (size_t)(m0 + lane) * D_ + k + 32, 0, 1);
      __builtin_prefetch(BT + (size_t)(n0 + lane) * D_ + k + 32, 0, 1);
    }
    v16bf a0 = load_a_frag(A, D_, m0,      k);
    v16bf a1 = load_a_frag(A, D_, m0 + 16, k);
#pragma unroll
    for (int ni = 0; ni < 4; ++ni) {
      v16bf bfr = load_b_frag(BT, D_, n0 + ni * 16, k);
      acc[0][ni] = wmma_bf16(a0, bfr, acc[0][ni]);
      acc[1][ni] = wmma_bf16(a1, bfr, acc[1][ni]);
    }
  }

  bf16* Cp = out + (size_t)z * SQ_ * DH_;
  const int nl = lane & 15;
  const int rb = (lane >> 4) << 3;
  if (!transposed) {
#pragma unroll
    for (int mi = 0; mi < 2; ++mi)
#pragma unroll
      for (int ni = 0; ni < 4; ++ni) {
        const int nn = n0 + ni * 16 + nl;
        const int mb = m0 + mi * 16 + rb;
#pragma unroll
        for (int i = 0; i < 8; ++i)
          Cp[(size_t)(mb + i) * DH_ + nn] = f2bf(acc[mi][ni][i]);
      }
  } else {  // out layout [DH][S]: per-lane 8 contiguous elems along m
#pragma unroll
    for (int mi = 0; mi < 2; ++mi)
#pragma unroll
      for (int ni = 0; ni < 4; ++ni) {
        const int nn = n0 + ni * 16 + nl;
        const int mb = m0 + mi * 16 + rb;
        v8bf pk;
#pragma unroll
        for (int i = 0; i < 8; ++i) pk[i] = f2bf(acc[mi][ni][i]);
        *(v8bf*)(Cp + (size_t)nn * SQ_ + mb) = pk;
      }
  }
}

// ---------------------------------------------------------------------------
// Flash attention with cooperative async K/V staging through LDS.
// Per (b,h): Q [SQ][DH] bf16, K [SKV][DH] bf16, VT [DH][SKV] bf16
//   -> cat bf16 [B][SQ][H*DH].
// grid (SQ/128, B*H), block 256: each wave owns 16 Q rows; the block shares
// each 64-row KV tile via LDS (loaded once with global_load_async_to_lds).
// ---------------------------------------------------------------------------
#define KLDS_ 136   // 64 rows x 128 bf16, padded (272B rows, 16B aligned)
#define VLDS_ 72    // 128 rows x 64 bf16, padded (144B rows, 16B aligned)

__global__ void __launch_bounds__(256)
flash_attn_kernel(const bf16* __restrict__ Q, const bf16* __restrict__ Km,
                  const bf16* __restrict__ VT, bf16* __restrict__ cat) {
  __shared__ bf16 sK[64 * KLDS_];       // ~17 KB
  __shared__ bf16 sV[128 * VLDS_];      // ~18 KB
  __shared__ bf16 sP[8 * 16 * 64];      // 16 KB, per-wave P tiles

  const int lane = threadIdx.x & 31;
  const int w    = threadIdx.x >> 5;
  const int z = blockIdx.y, b = z / H_, h = z % H_;
  const int qm0 = blockIdx.x * 128 + w * 16;

  const bf16* Qp = Q  + (size_t)z * SQ_  * DH_;
  const bf16* Kp = Km + (size_t)z * SKV_ * DH_;
  const bf16* Vp = VT + (size_t)z * DH_  * SKV_;

  // Q fragments for the full DH=128 contraction, kept resident
  v16bf qf[4];
#pragma unroll
  for (int kk = 0; kk < 4; ++kk) qf[kk] = load_a_frag(Qp, DH_, qm0, kk * 32);

  v8f rowmax, rowsum, O[8];
#pragma unroll
  for (int i = 0; i < 8; ++i) { rowmax[i] = -3.0e38f; rowsum[i] = 0.0f; }
#pragma unroll
  for (int nb = 0; nb < 8; ++nb) O[nb] = v8zero();

  const float scale = 0.08838834764831845f;  // 1/sqrt(128)
  bf16* myP = sP + w * (16 * 64);
  const int nl = lane & 15;
  const int rb = (lane >> 4) << 3;

  for (int kv0 = 0; kv0 < SKV_; kv0 += 64) {
    // ---- stage K tile (64x128) and V tile (128x64) into LDS ----
#pragma unroll
    for (int c = 0; c < 4; ++c) {             // 1024 16B chunks / 256 threads
      int idx = threadIdx.x + c * 256;
      int kr = idx >> 4, kc = idx & 15;       // K: 16 chunks per 256B row
      cp_g2l_16B(Kp + (size_t)(kv0 + kr) * DH_ + kc * 8, sK + kr * KLDS_ + kc * 8);
      int vr = idx >> 3, vc = idx & 7;        // V: 8 chunks per 128B row
      cp_g2l_16B(Vp + (size_t)vr * SKV_ + kv0 + vc * 8, sV + vr * VLDS_ + vc * 8);
    }
    wait_async_lds();
    __syncthreads();

    // ---- S = scale * Q K^T for this 16x64 tile (K from LDS) ----
    v8f s[4];
#pragma unroll
    for (int n = 0; n < 4; ++n) {
      v8f a = v8zero();
#pragma unroll
      for (int kk = 0; kk < 4; ++kk) {
        v16bf kb = load_b_frag(sK, KLDS_, n * 16, kk * 32);
        a = wmma_bf16(qf[kk], kb, a);
      }
#pragma unroll
      for (int i = 0; i < 8; ++i) a[i] *= scale;
      s[n] = a;
    }

    // ---- online softmax (rows live in C-layout vector elements) ----
    v8f tmax = s[0];
#pragma unroll
    for (int n = 1; n < 4; ++n)
#pragma unroll
      for (int i = 0; i < 8; ++i) tmax[i] = fmaxf(tmax[i], s[n][i]);
#pragma unroll
    for (int m = 1; m < 16; m <<= 1)
#pragma unroll
      for (int i = 0; i < 8; ++i)
        tmax[i] = fmaxf(tmax[i], __shfl_xor(tmax[i], m, 32));

    v8f nmax, alpha;
#pragma unroll
    for (int i = 0; i < 8; ++i) {
      nmax[i]  = fmaxf(rowmax[i], tmax[i]);
      alpha[i] = __expf(rowmax[i] - nmax[i]);
    }
    v8f tsum = v8zero();
#pragma unroll
    for (int n = 0; n < 4; ++n)
#pragma unroll
      for (int i = 0; i < 8; ++i) {
        float p = __expf(s[n][i] - nmax[i]);
        s[n][i] = p;
        tsum[i] += p;
      }
#pragma unroll
    for (int m = 1; m < 16; m <<= 1)
#pragma unroll
      for (int i = 0; i < 8; ++i) tsum[i] += __shfl_xor(tsum[i], m, 32);
#pragma unroll
    for (int i = 0; i < 8; ++i) {
      rowsum[i] = rowsum[i] * alpha[i] + tsum[i];
      rowmax[i] = nmax[i];
    }
#pragma unroll
    for (int nb = 0; nb < 8; ++nb)
#pragma unroll
      for (int i = 0; i < 8; ++i) O[nb][i] *= alpha[i];

    // ---- C-layout P -> per-wave LDS tile -> A-fragment layout ----
#pragma unroll
    for (int n = 0; n < 4; ++n)
#pragma unroll
      for (int i = 0; i < 8; ++i)
        myP[(rb + i) * 64 + n * 16 + nl] = f2bf(s[n][i]);
    // same-wave DS ops stay in order; no cross-wave readers of myP
    v16bf pa0 = load_a_frag(myP, 64, 0, 0);
    v16bf pa1 = load_a_frag(myP, 64, 0, 32);

    // ---- O += P @ V (V tile in LDS, rows = dh) ----
#pragma unroll
    for (int nb = 0; nb < 8; ++nb) {
      v16bf vb0 = load_b_frag(sV, VLDS_, nb * 16, 0);
      v16bf vb1 = load_b_frag(sV, VLDS_, nb * 16, 32);
      O[nb] = wmma_bf16(pa0, vb0, O[nb]);
      O[nb] = wmma_bf16(pa1, vb1, O[nb]);
    }
    __syncthreads();   // everyone done with sK/sV before next staging
  }

  // ---- normalize and write head-concat bf16 output ----
  v8f inv;
#pragma unroll
  for (int i = 0; i < 8; ++i) inv[i] = 1.0f / rowsum[i];
#pragma unroll
  for (int nb = 0; nb < 8; ++nb)
#pragma unroll
    for (int i = 0; i < 8; ++i) {
      size_t row = (size_t)b * SQ_ + qm0 + rb + i;
      cat[row * HD_ + h * DH_ + nb * 16 + nl] = f2bf(O[nb][i] * inv[i]);
    }
}

// ---------------------------------------------------------------------------
// Big flat GEMM with epilogue: out = [relu](A @ BT^T + bias) + res  (f32 out)
// A bf16 [M][K], BT bf16 [N][K], bias f32 [N], res f32 [M][N]
// grid (M/128, N/128), block 256: 8 waves as 4(M)x2(N), 32x64 per wave
// ---------------------------------------------------------------------------
__global__ void __launch_bounds__(256)
gemm_ep_kernel(const bf16* __restrict__ A, const bf16* __restrict__ BT,
               const float* __restrict__ bias, const float* __restrict__ res,
               float* __restrict__ out, int M, int N, int K, int relu) {
  const int lane = threadIdx.x & 31;
  const int w    = threadIdx.x >> 5;
  const int wm = w & 3, wn = w >> 2;
  const int m0 = blockIdx.x * 128 + wm * 32;
  const int n0 = blockIdx.y * 128 + wn * 64;

  v8f acc[2][4];
#pragma unroll
  for (int i = 0; i < 2; ++i)
#pragma unroll
    for (int j = 0; j < 4; ++j) acc[i][j] = v8zero();

  for (int k = 0; k < K; k += 32) {
    if (k + 32 < K) {
      __builtin_prefetch(A  + (size_t)(m0 + lane) * K + k + 32, 0, 1);
      __builtin_prefetch(BT + (size_t)(n0 + lane) * K + k + 32, 0, 1);
    }
    v16bf a0 = load_a_frag(A, K, m0,      k);
    v16bf a1 = load_a_frag(A, K, m0 + 16, k);
#pragma unroll
    for (int ni = 0; ni < 4; ++ni) {
      v16bf bfr = load_b_frag(BT, K, n0 + ni * 16, k);
      acc[0][ni] = wmma_bf16(a0, bfr, acc[0][ni]);
      acc[1][ni] = wmma_bf16(a1, bfr, acc[1][ni]);
    }
  }

  const int nl = lane & 15;
  const int rb = (lane >> 4) << 3;
#pragma unroll
  for (int mi = 0; mi < 2; ++mi)
#pragma unroll
    for (int ni = 0; ni < 4; ++ni) {
      const int nn = n0 + ni * 16 + nl;
      const int mb = m0 + mi * 16 + rb;
      const float bv = bias[nn];
#pragma unroll
      for (int i = 0; i < 8; ++i) {
        size_t idx = (size_t)(mb + i) * N + nn;
        float t = acc[mi][ni][i] + bv;
        if (relu) t = fmaxf(t, 0.0f);
        out[idx] = t + res[idx];
      }
    }
}

// ---------------------------------------------------------------------------
// Row LayerNorm over N=1024 (one row per 256-thread block, in-place safe).
// Optionally emits a bf16 copy for the next GEMM's A matrix.
// ---------------------------------------------------------------------------
__global__ void __launch_bounds__(256)
layernorm_kernel(const float* __restrict__ x, const float* __restrict__ g,
                 const float* __restrict__ bb, float* __restrict__ outf,
                 bf16* __restrict__ outb, int N) {
  __shared__ float red[2][8];
  const size_t row = blockIdx.x;
  const float* xr = x + row * N;
  float v[4], s = 0.0f, s2 = 0.0f;
#pragma unroll
  for (int i = 0; i < 4; ++i) {
    float t = xr[threadIdx.x + i * 256];
    v[i] = t; s += t; s2 += t * t;
  }
#pragma unroll
  for (int m = 1; m < 32; m <<= 1) {
    s  += __shfl_xor(s,  m, 32);
    s2 += __shfl_xor(s2, m, 32);
  }
  const int w = threadIdx.x >> 5;
  if ((threadIdx.x & 31) == 0) { red[0][w] = s; red[1][w] = s2; }
  __syncthreads();
  s = 0.0f; s2 = 0.0f;
#pragma unroll
  for (int i = 0; i < 8; ++i) { s += red[0][i]; s2 += red[1][i]; }
  const float mu  = s / N;
  const float var = s2 / N - mu * mu;
  const float inv = rsqrtf(var + LN_EPS);
#pragma unroll
  for (int i = 0; i < 4; ++i) {
    int c = threadIdx.x + i * 256;
    float y = (v[i] - mu) * inv * g[c] + bb[c];
    outf[row * N + c] = y;
    if (outb) outb[row * N + c] = f2bf(y);
  }
}

// ---------------------------------------------------------------------------
// Host orchestration
// ---------------------------------------------------------------------------
extern "C" void kernel_launch(void* const* d_in, const int* in_sizes, int n_in,
                              void* d_out, int out_size, void* d_ws, size_t ws_size,
                              hipStream_t stream) {
  (void)in_sizes; (void)n_in; (void)out_size; (void)ws_size;

  const float* query = (const float*)d_in[0];
  const float* value = (const float*)d_in[1];
  const float* W1 = (const float*)d_in[2];
  const float* W2 = (const float*)d_in[3];
  const float* W3 = (const float*)d_in[4];
  const float* Wp = (const float*)d_in[5];
  const float* bp = (const float*)d_in[6];
  const float* Wt = (const float*)d_in[7];
  const float* bt = (const float*)d_in[8];
  const float* g1 = (const float*)d_in[9];
  const float* b1 = (const float*)d_in[10];
  const float* g2 = (const float*)d_in[11];
  const float* b2 = (const float*)d_in[12];

  char* ws = (char*)d_ws;
  const size_t MB = 1024ull * 1024ull;
  bf16*  qbf    = (bf16*)(ws + 0   * MB);   // 32 MB  query bf16
  bf16*  vbf    = (bf16*)(ws + 32  * MB);   // 32 MB  value bf16
  bf16*  W1T    = (bf16*)(ws + 64  * MB);   // 2 MB   [H][DH][D]
  bf16*  W2T    = (bf16*)(ws + 66  * MB);
  bf16*  W3T    = (bf16*)(ws + 68  * MB);
  bf16*  WpT    = (bf16*)(ws + 70  * MB);   // [P][HD]
  bf16*  WtT    = (bf16*)(ws + 72  * MB);   // [P][P]
  bf16*  qh     = (bf16*)(ws + 74  * MB);   // 32 MB  [B*H][SQ][DH]
  bf16*  kh     = (bf16*)(ws + 106 * MB);   // 32 MB  [B*H][SKV][DH]
  bf16*  vT     = (bf16*)(ws + 138 * MB);   // 32 MB  [B*H][DH][SKV]
  bf16*  cat    = (bf16*)(ws + 170 * MB);   // 32 MB  [B][SQ][HD]
  float* proj   = (float*)(ws + 202 * MB);  // 64 MB  f32
  bf16*  projbf = (bf16*)(ws + 74  * MB);   // reuse qh (attention done)
  float* traw   = (float*)(ws + 106 * MB);  // reuse kh+vT (64 MB)

  // 1) activations -> bf16
  cvt_kernel<<<4096, 256, 0, stream>>>(query, qbf, (size_t)B_ * SQ_ * D_);
  cvt_kernel<<<4096, 256, 0, stream>>>(value, vbf, (size_t)B_ * SKV_ * D_);

  // 2) weights -> bf16, transposed so B-fragments load contiguously
  tcvt_kernel<<<dim3(512, 1, H_), 256, 0, stream>>>(W1, W1T, D_, DH_);
  tcvt_kernel<<<dim3(512, 1, H_), 256, 0, stream>>>(W2, W2T, D_, DH_);
  tcvt_kernel<<<dim3(512, 1, H_), 256, 0, stream>>>(W3, W3T, D_, DH_);
  tcvt_kernel<<<dim3(4096, 1, 1), 256, 0, stream>>>(Wp, WpT, HD_, P_);
  tcvt_kernel<<<dim3(4096, 1, 1), 256, 0, stream>>>(Wt, WtT, P_, P_);

  // 3) per-head Q/K/V projections (V stored transposed for ctx GEMM)
  gemm_qkv_kernel<<<dim3(SQ_ / 128, DH_ / 128, B_ * H_), 256, 0, stream>>>(qbf, W1T, qh, 0);
  gemm_qkv_kernel<<<dim3(SKV_ / 128, DH_ / 128, B_ * H_), 256, 0, stream>>>(vbf, W2T, kh, 0);
  gemm_qkv_kernel<<<dim3(SKV_ / 128, DH_ / 128, B_ * H_), 256, 0, stream>>>(vbf, W3T, vT, 1);

  // 4) streaming-softmax attention -> concat(heads) bf16
  flash_attn_kernel<<<dim3(SQ_ / 128, B_ * H_), 256, 0, stream>>>(qh, kh, vT, cat);

  // 5) output projection + bias + residual(query), then LayerNorm (in place)
  gemm_ep_kernel<<<dim3((B_ * SQ_) / 128, P_ / 128), 256, 0, stream>>>(
      cat, WpT, bp, query, proj, B_ * SQ_, P_, HD_, 0);
  layernorm_kernel<<<B_ * SQ_, 256, 0, stream>>>(proj, g1, b1, proj, projbf, P_);

  // 6) relu FFN + residual(proj), then final LayerNorm -> d_out
  gemm_ep_kernel<<<dim3((B_ * SQ_) / 128, P_ / 128), 256, 0, stream>>>(
      projbf, WtT, bt, proj, traw, B_ * SQ_, P_, P_, 1);
  layernorm_kernel<<<B_ * SQ_, 256, 0, stream>>>(traw, g2, b2, (float*)d_out,
                                                 (bf16*)nullptr, P_);
}
